// HyperbolicKuramoto_28819230556792
// MI455X (gfx1250) — matrix operand order repaired and checked by reading, not compiled, not run on hardware
//
#include <hip/hip_runtime.h>
#include <hip/hip_bf16.h>
#include <math.h>

// ---------------------------------------------------------------------------
// HyperbolicKuramoto on MI455X (gfx1250, wave32)
//
// Per step:  (u + i v) = K @ z   (K real -> K@conj(z) = u - i v)
//            z' = z + dt*( -S*z^2/(2n) + omega*z + (K z)/(2n) ),  S = u - i v
//            clamp |z'| to 0.999
//
// Memory-bound on K. K constant across steps -> convert once to f16:
//   * halves the streamed bytes (128 MB/step instead of 256 MB)
//   * 128 MB K16 fits in the 192 MB L2 -> steps 2..128 run at L2 bandwidth
// GEMV done with v_wmma_f32_16x16x32_f16, B = [z_re, z_im, 0...0].
// Deterministic: no atomics, fixed partial-sum order, branch-free hot loop.
// ---------------------------------------------------------------------------

#define NN     8192
#define STEPS  128
#define CHUNKS 8
#define CHUNK  (NN / CHUNKS)   // 1024
#define TILES  (NN / 16)       // 512
#define CLAMPV 0.999f
#define INV2N  (1.0f / (2.0f * 8192.0f))

typedef __attribute__((ext_vector_type(16))) _Float16 v16h;
typedef __attribute__((ext_vector_type(8)))  _Float16 v8h;
typedef __attribute__((ext_vector_type(8)))  float    v8f;
typedef __attribute__((ext_vector_type(4)))  float    v4f;

// ---------------- K f32 -> f16 (once per call) ------------------------------
__global__ void __launch_bounds__(256) hk_convert_K(const float* __restrict__ K,
                                                    _Float16* __restrict__ K16) {
  size_t i = ((size_t)blockIdx.x * blockDim.x + threadIdx.x) * 8;
  v4f a = *(const v4f*)(K + i);
  v4f b = *(const v4f*)(K + i + 4);
  v8h h;
#pragma unroll
  for (int j = 0; j < 4; ++j) { h[j] = (_Float16)a[j]; h[j + 4] = (_Float16)b[j]; }
  *(v8h*)(K16 + i) = h;
}

// ---------------- init state from input z [N,2]; zero the zero-pad line ----
__global__ void __launch_bounds__(256) hk_init(const float* __restrict__ z,
                                               float* __restrict__ zre, float* __restrict__ zim,
                                               _Float16* __restrict__ zre16,
                                               _Float16* __restrict__ zim16,
                                               _Float16* __restrict__ zbuf, int w16) {
  int i = blockIdx.x * blockDim.x + threadIdx.x;
  float a = z[2 * i], b = z[2 * i + 1];
  zre[i] = a; zim[i] = b;
  if (w16) { zre16[i] = (_Float16)a; zim16[i] = (_Float16)b; }
  if (i < 32) zbuf[i] = (_Float16)0.0f;   // 64B zero line for B cols 2..15
}

// ---------------- WMMA f16 GEMV: (u,v) partials = K16 @ (zre,zim) -----------
// Wave w: row-tile = w>>3 (16 rows), k-chunk = w&7 (1024 k).
// A: 16x32 f16 tile of K16 (ISA 16-bit A layout: lanes 0-15 row M=n hold
//    k = k0+0..7 / k0+16..23; lanes 16-31 hold k0+8..15 / k0+24..31).
// B: 32x16 f16, col0 = zre, col1 = zim, cols 2..15 = 0 (lane n = column,
//    lanes >= 16 hold k rows 8..15 / 24..31). Lanes n>=2 read a shared 64B zero line
//    (offset masked to 0) so the hot loop is completely branch-free.
// C/D: 16x16 f32; col 0 (lanes 0,16) = u, col 1 (lanes 1,17) = v.
__global__ void __launch_bounds__(256) hk_gemv_wmma_f16(
    const _Float16* __restrict__ K16,
    const _Float16* __restrict__ zre16, const _Float16* __restrict__ zim16,
    const _Float16* __restrict__ zbuf,
    float* __restrict__ up, float* __restrict__ vp) {
  const int tid   = threadIdx.x;
  const int lane  = tid & 31;
  const int wv    = (blockIdx.x << 3) | (tid >> 5);
  const int tile  = wv >> 3;
  const int chunk = wv & 7;
  const int n     = lane & 15;
  const int half  = lane >> 4;

  const _Float16* arow = K16 + (size_t)(tile * 16 + n) * (size_t)NN + half * 8;
  // Per-lane B source: col0 -> zre, col1 -> zim, cols 2..15 -> zero line.
  const _Float16* bcol = (n == 0) ? zre16 : ((n == 1) ? zim16 : zbuf);
  const int koffmask = (n < 2) ? ~0 : 0;   // lanes >=2 always reload offset 0
  const int kbase = chunk * CHUNK;

  v8f c = {};
#pragma unroll 4
  for (int it = 0; it < CHUNK / 32; ++it) {
    const int k0 = kbase + it * 32;
    const _Float16* ap = arow + k0;
    v8h a0 = *(const v8h*)ap;            // k0 + half*8 + [0..7]
    v8h a1 = *(const v8h*)(ap + 16);     // k0 + half*8 + [16..23]
    __builtin_prefetch(ap + 256, 0, 0);  // stream 512B ahead in k

    const int boff = (k0 + half * 8) & koffmask;
    v8h b0 = *(const v8h*)(bcol + boff);       // k rows half*8 + 0..7
    v8h b1 = *(const v8h*)(bcol + boff + 16);  // k rows half*8 + 16..23

    v16h A, B;
#pragma unroll
    for (int j = 0; j < 8; ++j) { A[j] = a0[j]; A[j + 8] = a1[j];
                                  B[j] = b0[j]; B[j + 8] = b1[j]; }

    c = __builtin_amdgcn_wmma_f32_16x16x32_f16(
        /*neg_a=*/false, A, /*neg_b=*/false, B,
        /*c_mod=*/(short)0, c, /*reuse_a=*/false, /*reuse_b=*/false);
  }

  // C/D 16x16 f32 layout: lane = N-col; VGPR r -> M=r (lanes<16) / M=r+8.
  if ((lane & 14) == 0) {              // lanes 0,1,16,17
    float* dst = (lane & 1) ? (vp + chunk * NN) : (up + chunk * NN);
    const int rbase = tile * 16 + half * 8;
#pragma unroll
    for (int r = 0; r < 8; ++r) dst[rbase + r] = c[r];
  }
}

// ---------------- fallback f32 GEMV (wave per row, shuffle reduce) ----------
__global__ void __launch_bounds__(256) hk_gemv_f32(
    const float* __restrict__ K,
    const float* __restrict__ zre, const float* __restrict__ zim,
    float* __restrict__ u, float* __restrict__ v) {
  const int lane = threadIdx.x & 31;
  const int row  = (blockIdx.x << 3) | (threadIdx.x >> 5);
  const float* kp = K + (size_t)row * (size_t)NN;
  float au = 0.f, av = 0.f;
  for (int k = lane * 4; k < NN; k += 128) {
    v4f kk = *(const v4f*)(kp + k);
    v4f ar = *(const v4f*)(zre + k);
    v4f ai = *(const v4f*)(zim + k);
#pragma unroll
    for (int j = 0; j < 4; ++j) { au += kk[j] * ar[j]; av += kk[j] * ai[j]; }
  }
#pragma unroll
  for (int off = 16; off > 0; off >>= 1) {
    au += __shfl_down(au, off, 32);
    av += __shfl_down(av, off, 32);
  }
  if (lane == 0) { u[row] = au; v[row] = av; }
}

// ---------------- elementwise step update + clamp ---------------------------
__global__ void __launch_bounds__(256) hk_step_update(
    const float* __restrict__ up, const float* __restrict__ vp, int nchunks,
    float* __restrict__ zre, float* __restrict__ zim,
    _Float16* __restrict__ zre16, _Float16* __restrict__ zim16, int w16,
    const float* __restrict__ pomega, const float* __restrict__ pdt) {
  const int i = blockIdx.x * blockDim.x + threadIdx.x;
  const float omega = pomega[0];
  const float dt    = pdt[0];

  float u = 0.f, v = 0.f;
  for (int cth = 0; cth < nchunks; ++cth) {   // fixed order -> deterministic
    u += up[cth * NN + i];
    v += vp[cth * NN + i];
  }

  float a = zre[i], b = zim[i];
  // S = u - i v ; z^2 = (a^2 - b^2) + i(2ab)
  const float x = a * a - b * b;
  const float y = 2.0f * a * b;
  const float t1r = -INV2N * (u * x + v * y);
  const float t1i = -INV2N * (u * y - v * x);
  float nr = a + dt * (t1r + omega * a + INV2N * u);
  float ni = b + dt * (t1i + omega * b + INV2N * v);

  const float m = sqrtf(nr * nr + ni * ni);
  if (m >= CLAMPV) { const float s = CLAMPV / m; nr *= s; ni *= s; }

  zre[i] = nr; zim[i] = ni;
  if (w16) { zre16[i] = (_Float16)nr; zim16[i] = (_Float16)ni; }
}

// ---------------- final clamp + pack [N,2] ----------------------------------
__global__ void __launch_bounds__(256) hk_finalize(const float* __restrict__ zre,
                                                   const float* __restrict__ zim,
                                                   float* __restrict__ out) {
  const int i = blockIdx.x * blockDim.x + threadIdx.x;
  float a = zre[i], b = zim[i];
  const float m = sqrtf(a * a + b * b);
  if (m >= CLAMPV) { const float s = CLAMPV / m; a *= s; b *= s; }
  out[2 * i]     = a;
  out[2 * i + 1] = b;
}

// ---------------------------------------------------------------------------
extern "C" void kernel_launch(void* const* d_in, const int* in_sizes, int n_in,
                              void* d_out, int out_size, void* d_ws, size_t ws_size,
                              hipStream_t stream) {
  const float* z     = (const float*)d_in[0];   // [N,2]
  const float* K     = (const float*)d_in[1];   // [N,N]
  const float* omega = (const float*)d_in[2];   // scalar
  const float* dt    = (const float*)d_in[3];   // scalar
  (void)in_sizes; (void)n_in; (void)out_size;

  // Workspace layout (bytes):
  const size_t OFF_ZRE   = 0;
  const size_t OFF_ZIM   = OFF_ZRE + (size_t)NN * 4;
  const size_t OFF_ZRE16 = OFF_ZIM + (size_t)NN * 4;
  const size_t OFF_ZIM16 = OFF_ZRE16 + (size_t)NN * 2;
  const size_t OFF_ZBUF  = OFF_ZIM16 + (size_t)NN * 2;          // 64B zero line
  const size_t OFF_UP    = 98560;                               // 256-aligned
  const size_t OFF_VP    = OFF_UP + (size_t)CHUNKS * NN * 4;
  const size_t OFF_K16   = OFF_VP + (size_t)CHUNKS * NN * 4;
  const size_t NEED_F16  = OFF_K16 + (size_t)NN * NN * 2;       // ~134.8 MB

  char* w = (char*)d_ws;
  float*    zre   = (float*)(w + OFF_ZRE);
  float*    zim   = (float*)(w + OFF_ZIM);
  _Float16* zre16 = (_Float16*)(w + OFF_ZRE16);
  _Float16* zim16 = (_Float16*)(w + OFF_ZIM16);
  _Float16* zbuf  = (_Float16*)(w + OFF_ZBUF);
  float*    up    = (float*)(w + OFF_UP);
  float*    vp    = (float*)(w + OFF_VP);
  _Float16* K16   = (_Float16*)(w + OFF_K16);

  const bool f16path = (ws_size >= NEED_F16);
  const int  blkN    = NN / 256;   // 32 blocks of 256 for N-wide kernels

  hk_init<<<blkN, 256, 0, stream>>>(z, zre, zim, zre16, zim16, zbuf,
                                    f16path ? 1 : 0);

  if (f16path) {
    // 8192*8192 / (256*8) = 32768 blocks
    hk_convert_K<<<(NN * (size_t)NN) / (256 * 8), 256, 0, stream>>>(K, K16);
  }

  for (int s = 0; s < STEPS; ++s) {
    if (f16path) {
      // TILES*CHUNKS = 4096 waves, 8 waves/block -> 512 blocks
      hk_gemv_wmma_f16<<<(TILES * CHUNKS) / 8, 256, 0, stream>>>(K16, zre16, zim16,
                                                                 zbuf, up, vp);
      hk_step_update<<<blkN, 256, 0, stream>>>(up, vp, CHUNKS, zre, zim,
                                               zre16, zim16, 1, omega, dt);
    } else {
      // 8192 waves, 8 waves/block -> 1024 blocks; partials: single chunk
      hk_gemv_f32<<<NN / 8, 256, 0, stream>>>(K, zre, zim, up, vp);
      hk_step_update<<<blkN, 256, 0, stream>>>(up, vp, 1, zre, zim,
                                               zre16, zim16, 0, omega, dt);
    }
  }

  hk_finalize<<<blkN, 256, 0, stream>>>(zre, zim, (float*)d_out);
}
